// CausalSelfAttention_41738492183263
// MI455X (gfx1250) — matrix operand order repaired and checked by reading, not compiled, not run on hardware
//
#include <hip/hip_runtime.h>
#include <hip/hip_bf16.h>

// ---------------------------------------------------------------------------
// CDNA5 (gfx1250) causal self-attention: bf16 WMMA end-to-end.
//   pass 1: fp32 -> bf16 conversion of x / w_attn / w_proj
//   pass 2: qkv = x @ w_attn + b_attn   (WMMA GEMM, async-LDS pipelined)
//   pass 3: flash attention             (WMMA qk^T / p*v, DPP softmax,
//                                        double-buffered async KV pipeline)
//   pass 4: out = y @ w_proj + b_proj   (WMMA GEMM, fp32 out)
// ---------------------------------------------------------------------------

typedef __attribute__((ext_vector_type(16))) __bf16        v16bf;
typedef __attribute__((ext_vector_type(8)))  float         v8f;
typedef __attribute__((ext_vector_type(8)))  unsigned int  v8u;

#define WMMA_BF16(A_, B_, C_) \
  __builtin_amdgcn_wmma_f32_16x16x32_bf16(false, (A_), false, (B_), (short)0, (C_), false, false)

static __device__ __forceinline__ unsigned short f2bf(float f) {
  unsigned u = __builtin_bit_cast(unsigned, f);
  u += 0x7FFFu + ((u >> 16) & 1u);      // round-to-nearest-even
  return (unsigned short)(u >> 16);
}

// Low 32 bits of a flat pointer into LDS == byte offset in the workgroup's LDS.
static __device__ __forceinline__ unsigned lds_off(const void* p) {
  return (unsigned)(unsigned long long)p;
}

// CDNA5 async global->LDS copy, 16B per lane. Tracked by ASYNCcnt.
static __device__ __forceinline__ void async_load_b128(unsigned lds_byte_off,
                                                       const void* gptr) {
  asm volatile("global_load_async_to_lds_b128 %0, %1, off"
               :
               : "v"(lds_byte_off), "v"(gptr)
               : "memory");
}
static __device__ __forceinline__ void wait_async0() {
  asm volatile("s_wait_asynccnt 0" ::: "memory");
}

// DPP16 row_xmask butterfly within 16-lane groups (VALU, no LDS traffic).
template <int MASK>
static __device__ __forceinline__ float dpp_xor16(float x) {
  const int t = __builtin_amdgcn_update_dpp(0, __builtin_bit_cast(int, x),
                                            0x160 | MASK, 0xF, 0xF, true);
  return __builtin_bit_cast(float, t);
}
static __device__ __forceinline__ float red_max16(float x) {
  x = fmaxf(x, dpp_xor16<1>(x));
  x = fmaxf(x, dpp_xor16<2>(x));
  x = fmaxf(x, dpp_xor16<4>(x));
  x = fmaxf(x, dpp_xor16<8>(x));
  return x;
}
static __device__ __forceinline__ float red_sum16(float x) {
  x += dpp_xor16<1>(x);
  x += dpp_xor16<2>(x);
  x += dpp_xor16<4>(x);
  x += dpp_xor16<8>(x);
  return x;
}

// A-fragment (16x32 bf16, M x K). Lane L: row m = L&15; lanes 0-15 hold K 0-7 &
// 16-23, lanes 16-31 hold K 8-15 & 24-31, two K values packed per VGPR.
static __device__ __forceinline__ v16bf load_frag_a(const unsigned short* s, int lda, int lane) {
  const int m = lane & 15, half = lane >> 4;
  const unsigned short* r = s + m * lda + half * 8;
  v8u t;
#pragma unroll
  for (int v = 0; v < 8; ++v) {
    const int k = ((v & 3) * 2) + ((v >> 2) * 16);
    t[v] = *(const unsigned int*)(r + k);
  }
  return __builtin_bit_cast(v16bf, t);
}

// B-fragment (32x16 bf16, K x N) from an LDS tile stored TRANSPOSED as [n][k].
// Lane L: col n = L&15; lanes 0-15 hold K 0-7 & 16-23, lanes 16-31 hold K 8-15 & 24-31,
// two K values packed per VGPR (same K order as the A-fragment).
static __device__ __forceinline__ v16bf load_frag_b(const unsigned short* s, int ldb, int lane) {
  const int n = lane & 15, half = lane >> 4;
  const unsigned short* r = s + n * ldb + half * 8;
  v8u t;
#pragma unroll
  for (int v = 0; v < 8; ++v) t[v] = *(const unsigned int*)(r + ((v & 3) * 2) + ((v >> 2) * 16));
  return __builtin_bit_cast(v16bf, t);
}

// ---------------------------------------------------------------------------
// fp32 -> bf16 elementwise (float4 in, packed 2x bf16 out)
// ---------------------------------------------------------------------------
__global__ __launch_bounds__(256) void cvt_f32_to_bf16(const float* __restrict__ in,
                                                       unsigned short* __restrict__ out,
                                                       int n4) {
  const int i = blockIdx.x * 256 + threadIdx.x;
  if (i >= n4) return;
  const float4 v = ((const float4*)in)[i];
  ((unsigned int*)out)[2 * i + 0] = (unsigned)f2bf(v.x) | ((unsigned)f2bf(v.y) << 16);
  ((unsigned int*)out)[2 * i + 1] = (unsigned)f2bf(v.z) | ((unsigned)f2bf(v.w) << 16);
}

// ---------------------------------------------------------------------------
// bf16 GEMM: C[M,N] = A[M,K] @ B[K,N] + bias[N]
// Block tile 128x128, K-step 32. 8 waves as 2(M) x 4(N); wave tile 64x32
// (4x2 WMMA tiles -> 8 v_wmma per K-step, 6 fragment loads).
// Double-buffered LDS; A staged via async global->LDS (B128), B register-
// prefetched and stored transposed while WMMAs execute. One barrier/K-step.
// A rows padded to 144B (16B-aligned async stores, 36-bank step, conflict-free).
// ---------------------------------------------------------------------------
template <bool OUT_F32>
__global__ __launch_bounds__(256) void gemm_bf16_kernel(const unsigned short* __restrict__ A,
                                                        const unsigned short* __restrict__ Bm,
                                                        const float* __restrict__ bias,
                                                        void* __restrict__ Cout,
                                                        int M, int N, int K) {
  constexpr int BM = 128, BN = 128, BK = 32;
  constexpr int LDA = 72;  // halfwords; 144B rows
  constexpr int LDB = 34;  // halfwords; [n][k] transposed tile
  __shared__ __align__(16) unsigned short sA[2][BM * LDA];
  __shared__ __align__(16) unsigned short sBt[2][BN * LDB];

  const int tid = threadIdx.x, lane = tid & 31, wave = tid >> 5;
  const int wm = wave & 1, wn = wave >> 1;
  const int bm = blockIdx.x * BM, bn = blockIdx.y * BN;
  const unsigned sA_base = lds_off(&sA[0][0]);
  constexpr unsigned ABUF = BM * LDA * 2;  // bytes per A buffer

  v8f acc[4][2];
#pragma unroll
  for (int mt = 0; mt < 4; ++mt)
#pragma unroll
    for (int nt = 0; nt < 2; ++nt) acc[mt][nt] = {};

  uint4 breg[2];

  auto stage_A = [&](int k0, int pb) {  // 128x32 bf16, 2 async B128 per thread
    int c = tid;
#pragma unroll
    for (int it = 0; it < 2; ++it, c += 256) {
      const int row = c >> 2, cc = (c & 3) * 8;
      async_load_b128(sA_base + (unsigned)pb * ABUF + (unsigned)(row * LDA + cc) * 2u,
                      A + (size_t)(bm + row) * K + k0 + cc);
    }
  };
  auto load_B = [&](int k0) {           // 32x128 bf16 into registers
    int c = tid;
#pragma unroll
    for (int it = 0; it < 2; ++it, c += 256) {
      const int row = c >> 4, nn = (c & 15) * 8;
      breg[it] = *(const uint4*)(Bm + (size_t)(k0 + row) * N + bn + nn);
    }
  };
  auto store_B = [&](int pb) {          // transpose into sBt[pb][n][k]
    int c = tid;
#pragma unroll
    for (int it = 0; it < 2; ++it, c += 256) {
      const int row = c >> 4, nn = (c & 15) * 8;
      const unsigned int w[4] = {breg[it].x, breg[it].y, breg[it].z, breg[it].w};
#pragma unroll
      for (int e = 0; e < 8; ++e)
        sBt[pb][(nn + e) * LDB + row] = (unsigned short)(w[e >> 1] >> ((e & 1) * 16));
    }
  };

  // prologue: fill buffer 0
  stage_A(0, 0);
  load_B(0);
  store_B(0);
  wait_async0();
  __syncthreads();

  int p = 0;
  for (int k0 = 0; k0 < K; k0 += BK) {
    const bool next = (k0 + BK) < K;
    if (next) { stage_A(k0 + BK, p ^ 1); load_B(k0 + BK); }

    v16bf bf0 = load_frag_b(&sBt[p][(wn * 32 + 0)  * LDB], LDB, lane);
    v16bf bf1 = load_frag_b(&sBt[p][(wn * 32 + 16) * LDB], LDB, lane);
#pragma unroll
    for (int mt = 0; mt < 4; ++mt) {
      const v16bf af = load_frag_a(&sA[p][(wm * 64 + mt * 16) * LDA], LDA, lane);
      acc[mt][0] = WMMA_BF16(af, bf0, acc[mt][0]);
      acc[mt][1] = WMMA_BF16(af, bf1, acc[mt][1]);
    }

    if (next) store_B(p ^ 1);
    wait_async0();        // own async writes into buf p^1 have landed
    __syncthreads();      // everyone done reading buf p / writing buf p^1
    p ^= 1;
  }

  // epilogue: C/D layout -> lane holds col n = lane&15, rows (lane>>4)*8 + r
  const int n_in = lane & 15, half = lane >> 4;
#pragma unroll
  for (int mt = 0; mt < 4; ++mt) {
#pragma unroll
    for (int nt = 0; nt < 2; ++nt) {
      const int col = bn + wn * 32 + nt * 16 + n_in;
      const float bv = bias[col];
#pragma unroll
      for (int r = 0; r < 8; ++r) {
        const int row = bm + wm * 64 + mt * 16 + half * 8 + r;
        const float val = acc[mt][nt][r] + bv;
        if constexpr (OUT_F32)
          ((float*)Cout)[(size_t)row * N + col] = val;
        else
          ((unsigned short*)Cout)[(size_t)row * N + col] = f2bf(val);
      }
    }
  }
}

// ---------------------------------------------------------------------------
// Flash attention. Block = (b, h, 128-row q tile), 8 waves, each wave owns a
// 16-row q strip (q held as two A-fragments). 32-wide KV tiles, software-
// pipelined with double-buffered LDS: while tile j is computed, tile j+1's
// K flows via async global->LDS and V sits in registers; one barrier/iter.
//   S(16x32) = q @ k^T  : 4 WMMA   softmax: DPP16 row_xmask (VALU only)
//   P -> per-wave LDS (s_wait_dscnt), O(16x64) += P @ V : 4 WMMA
// ---------------------------------------------------------------------------
__global__ __launch_bounds__(256) void flash_attn_kernel(const unsigned short* __restrict__ qkv,
                                                         unsigned short* __restrict__ Y) {
  constexpr int T = 2048, C = 768, C3 = 3 * C, HS = 64;
  constexpr int LDQ = 72, LDK = 72, LDV = 34, LDP = 34;
  const int qt = blockIdx.x, h = blockIdx.y, b = blockIdx.z;
  const int tid = threadIdx.x, lane = tid & 31, wave = tid >> 5;
  const int n_in = lane & 15, half = lane >> 4;

  __shared__ __align__(16) unsigned short sQ[128 * LDQ];        // [qrow][hs]
  __shared__ __align__(16) unsigned short sK[2][32 * LDK];      // [kv][hs]
  __shared__ __align__(16) unsigned short sVt[2][64 * LDV];     // [hs][kv]
  __shared__ __align__(16) unsigned short sP[8 * 16 * LDP];     // per-wave P

  const unsigned sQ_base = lds_off(&sQ[0]);
  const unsigned sK_base = lds_off(&sK[0][0]);
  constexpr unsigned KBUF = 32 * LDK * 2;  // bytes per K buffer

  // per-thread staging coordinates (32x64 tile: 1 x B128 per thread)
  const int srow = tid >> 3, se0 = (tid & 7) * 8;
  const unsigned skoff = (unsigned)(srow * LDK + se0) * 2u;
  const unsigned short* kptr = qkv + (size_t)(b * T + srow) * C3 + h * HS + C + se0;
  const unsigned short* vptr = kptr + C;   // v = k + C within the qkv row
  constexpr size_t KVSTEP = (size_t)32 * C3;

  uint4 vreg;
  auto store_V = [&](int pb) {  // transpose vreg into sVt[pb][hs][kv]
    const unsigned int wv[4] = {vreg.x, vreg.y, vreg.z, vreg.w};
#pragma unroll
    for (int e = 0; e < 8; ++e)
      sVt[pb][(se0 + e) * LDV + srow] = (unsigned short)(wv[e >> 1] >> ((e & 1) * 16));
  };

  // stage the 128x64 q tile (async global->LDS) and KV tile 0
  for (int c = tid; c < 1024; c += 256) {
    const int row = c >> 3, e0 = (c & 7) * 8;
    async_load_b128(sQ_base + (unsigned)(row * LDQ + e0) * 2u,
                    qkv + (size_t)(b * T + qt * 128 + row) * C3 + h * HS + e0);
  }
  async_load_b128(sK_base + skoff, kptr);
  vreg = *(const uint4*)vptr;
  store_V(0);
  wait_async0();
  __syncthreads();

  const v16bf qa0 = load_frag_a(&sQ[wave * 16 * LDQ + 0],  LDQ, lane);  // hs 0..31
  const v16bf qa1 = load_frag_a(&sQ[wave * 16 * LDQ + 32], LDQ, lane);  // hs 32..63

  const int qr0 = qt * 128 + wave * 16;
  float mrow[8], lrow[8];
#pragma unroll
  for (int r = 0; r < 8; ++r) { mrow[r] = -3.0e38f; lrow[r] = 0.f; }
  v8f o[4] = {{}, {}, {}, {}};

  const int njt = qt * 4 + 4;  // cover kv <= q-tile end (causal)
  int p = 0;
  for (int j = 0; j < njt; ++j) {
    const int kvbase = j * 32;
    const bool next = (j + 1) < njt;
    if (next) {  // tile j+1 in flight while we compute tile j
      kptr += KVSTEP;
      vptr += KVSTEP;
      async_load_b128(sK_base + (unsigned)(p ^ 1) * KBUF + skoff, kptr);
      vreg = *(const uint4*)vptr;
    }

    // S = q @ k^T over HS=64 (two WMMA K-steps per 16-col tile)
    const unsigned short* sk = &sK[p][0];
    v8f z = {};
    v8f s0 = WMMA_BF16(qa0, load_frag_b(sk + 0,  LDK, lane), z);
    s0     = WMMA_BF16(qa1, load_frag_b(sk + 32, LDK, lane), s0);
    v8f s1 = WMMA_BF16(qa0, load_frag_b(sk + 16 * LDK,      LDK, lane), z);
    s1     = WMMA_BF16(qa1, load_frag_b(sk + 16 * LDK + 32, LDK, lane), s1);

#pragma unroll
    for (int r = 0; r < 8; ++r) { s0[r] *= 0.125f; s1[r] *= 0.125f; }  // 1/sqrt(64)

    if (kvbase + 31 > qr0) {  // wave-uniform: only diagonal-crossing tiles mask
      const int kc0 = kvbase + n_in, kc1 = kvbase + 16 + n_in;
#pragma unroll
      for (int r = 0; r < 8; ++r) {
        const int qg = qr0 + half * 8 + r;
        if (kc0 > qg) s0[r] = -3.0e38f;
        if (kc1 > qg) s1[r] = -3.0e38f;
      }
    }

    float alpha[8];
#pragma unroll
    for (int r = 0; r < 8; ++r) {
      const float mx = red_max16(fmaxf(s0[r], s1[r]));
      const float mn = fmaxf(mrow[r], mx);
      alpha[r] = __expf(mrow[r] - mn);
      mrow[r] = mn;
      const float p0 = __expf(s0[r] - mn);
      const float p1 = __expf(s1[r] - mn);
      s0[r] = p0; s1[r] = p1;
      lrow[r] = lrow[r] * alpha[r] + red_sum16(p0 + p1);
    }
#pragma unroll
    for (int cc = 0; cc < 4; ++cc)
#pragma unroll
      for (int r = 0; r < 8; ++r) o[cc][r] *= alpha[r];

    // transpose P through per-wave LDS scratch into A-fragment layout
    unsigned short* sp = &sP[wave * 16 * LDP];
#pragma unroll
    for (int r = 0; r < 8; ++r) {
      sp[(half * 8 + r) * LDP + n_in]      = f2bf(s0[r]);
      sp[(half * 8 + r) * LDP + 16 + n_in] = f2bf(s1[r]);
    }
    asm volatile("s_wait_dscnt 0" ::: "memory");  // wave-local LDS RAW fence
    const v16bf pa = load_frag_a(sp, LDP, lane);
    const unsigned short* sv = &sVt[p][0];
    o[0] = WMMA_BF16(pa, load_frag_b(sv + 0,        LDV, lane), o[0]);
    o[1] = WMMA_BF16(pa, load_frag_b(sv + 16 * LDV, LDV, lane), o[1]);
    o[2] = WMMA_BF16(pa, load_frag_b(sv + 32 * LDV, LDV, lane), o[2]);
    o[3] = WMMA_BF16(pa, load_frag_b(sv + 48 * LDV, LDV, lane), o[3]);

    if (next) store_V(p ^ 1);  // V(j+1) regs -> other buffer
    wait_async0();             // our K(j+1) async writes have landed
    __syncthreads();           // all waves done with buffer p
    p ^= 1;
  }

  // normalize and store y[b, t, h*HS + n] as bf16
#pragma unroll
  for (int r = 0; r < 8; ++r) lrow[r] = 1.f / lrow[r];
#pragma unroll
  for (int cc = 0; cc < 4; ++cc) {
#pragma unroll
    for (int r = 0; r < 8; ++r) {
      const int row = qr0 + half * 8 + r;
      const int col = h * HS + cc * 16 + n_in;
      Y[(size_t)(b * T + row) * C + col] = f2bf(o[cc][r] * lrow[r]);
    }
  }
}

// ---------------------------------------------------------------------------
extern "C" void kernel_launch(void* const* d_in, const int* in_sizes, int n_in,
                              void* d_out, int out_size, void* d_ws, size_t ws_size,
                              hipStream_t stream) {
  (void)in_sizes; (void)n_in; (void)out_size; (void)ws_size;
  const float* x      = (const float*)d_in[0];
  const float* w_attn = (const float*)d_in[1];
  const float* b_attn = (const float*)d_in[2];
  const float* w_proj = (const float*)d_in[3];
  const float* b_proj = (const float*)d_in[4];

  constexpr int Bc = 4, T = 2048, C = 768, NH = 12;
  constexpr int M = Bc * T;      // 8192
  constexpr int N1 = 3 * C;      // 2304

  // workspace layout (bf16 halfwords), all 16B-aligned
  unsigned short* xb  = (unsigned short*)d_ws;          // [M,C]
  unsigned short* wab = xb  + (size_t)M * C;            // [C,3C]
  unsigned short* wpb = wab + (size_t)C * N1;           // [C,C]
  unsigned short* qkv = wpb + (size_t)C * C;            // [M,3C]
  unsigned short* yb  = qkv + (size_t)M * N1;           // [M,C]

  cvt_f32_to_bf16<<<dim3((M * C / 4 + 255) / 256),  dim3(256), 0, stream>>>(x,      xb,  M * C / 4);
  cvt_f32_to_bf16<<<dim3((C * N1 / 4 + 255) / 256), dim3(256), 0, stream>>>(w_attn, wab, C * N1 / 4);
  cvt_f32_to_bf16<<<dim3((C * C / 4 + 255) / 256),  dim3(256), 0, stream>>>(w_proj, wpb, C * C / 4);

  gemm_bf16_kernel<false><<<dim3(M / 128, N1 / 128), dim3(256), 0, stream>>>(
      xb, wab, b_attn, (void*)qkv, M, N1, C);

  flash_attn_kernel<<<dim3(T / 128, NH, Bc), dim3(256), 0, stream>>>(qkv, yb);

  gemm_bf16_kernel<true><<<dim3(M / 128, C / 128), dim3(256), 0, stream>>>(
      yb, wpb, b_proj, d_out, M, C, C);
}
